// SLGT2LocalAttention_71889162600607
// MI455X (gfx1250) — matrix operand order repaired and checked by reading, not compiled, not run on hardware
//
#include <hip/hip_runtime.h>

#define DEV __device__ __forceinline__

typedef __attribute__((ext_vector_type(16))) __bf16 v16bf;
typedef __attribute__((ext_vector_type(8)))  float  v8f;

#if defined(__has_builtin)
#if __has_builtin(__builtin_amdgcn_tensor_load_to_lds) && __has_builtin(__builtin_amdgcn_s_wait_tensorcnt)
#define USE_TDM 1
#endif
#endif

#ifdef USE_TDM
typedef unsigned int u32x4 __attribute__((ext_vector_type(4)));
typedef int          i32x8 __attribute__((ext_vector_type(8)));
typedef int          i32x4 __attribute__((ext_vector_type(4)));
#endif

// ---- problem constants (from reference) ----
static constexpr int Bq = 2, Dd = 8, Hh = 64, Ww = 64, Cc = 256;
static constexpr int Nn = Dd * Hh * Ww;   // 32768
static constexpr int BN = Bq * Nn;        // 65536
static constexpr int GDIM = 128, GHEADS = 4, HD = 32;

DEV unsigned short f2bf(float f) {
    unsigned int u = __float_as_uint(f);
    u += 0x7FFFu + ((u >> 16) & 1u);      // round-to-nearest-even
    return (unsigned short)(u >> 16);
}

union FragU { v16bf v; unsigned short s[16]; };

// A fragment: 16x32 bf16, element (m,k) at lds[(row_base+m)*stride + k_base + k]
// Contiguous k order used here: lane = m + ((k>>4)&1)*16 ; vgpr = (k&15)>>1 ; half = k&1
DEV v16bf load_a_frag(const unsigned short* lds, int row_base, int k_base, int stride, int lane) {
    FragU u;
    int m = lane & 15, half = lane >> 4;
    const unsigned short* p = lds + (row_base + m) * stride + k_base;
#pragma unroll
    for (int g = 0; g < 8; ++g) {
        int k = (half << 4) + (g << 1);
        u.s[2 * g]     = p[k];
        u.s[2 * g + 1] = p[k + 1];
    }
    return u.v;
}

// Same A layout but source is fp32 in LDS (converted during gather)
DEV v16bf load_a_frag_f32(const float* lds, int row_base, int k_base, int stride, int lane) {
    FragU u;
    int m = lane & 15, half = lane >> 4;
    const float* p = lds + (row_base + m) * stride + k_base;
#pragma unroll
    for (int g = 0; g < 8; ++g) {
        int k = (half << 4) + (g << 1);
        u.s[2 * g]     = f2bf(p[k]);
        u.s[2 * g + 1] = f2bf(p[k + 1]);
    }
    return u.v;
}

// B fragment: 32x16 bf16, element (k,n) at lds[(k_base+k)*stride + col + n]
// Contiguous k order used here: lane = ((k>>4)<<4)|n ; vgpr = (k&15)>>1 ; half = k&1
DEV v16bf load_b_frag(const unsigned short* lds, int k_base, int col, int stride, int lane) {
    FragU u;
    int n = lane & 15, khalf = lane >> 4;
    const unsigned short* p = lds + col + n;
#pragma unroll
    for (int g = 0; g < 8; ++g) {
        int k = k_base + (khalf << 4) + (g << 1);
        u.s[2 * g]     = p[k * stride];
        u.s[2 * g + 1] = p[(k + 1) * stride];
    }
    return u.v;
}

// B fragment gathered from transposed storage: B(k,n) = lds[(row_base+n)*stride + k_base + k]
DEV v16bf load_bT_frag(const unsigned short* lds, int row_base, int k_base, int stride, int lane) {
    FragU u;
    int n = lane & 15, khalf = lane >> 4;
    const unsigned short* p = lds + (row_base + n) * stride + k_base;
#pragma unroll
    for (int g = 0; g < 8; ++g) {
        int k = (khalf << 4) + (g << 1);
        u.s[2 * g]     = p[k];
        u.s[2 * g + 1] = p[k + 1];
    }
    return u.v;
}

DEV v8f wmma_bf16(v16bf a, v16bf b, v8f c) {
    return __builtin_amdgcn_wmma_f32_16x16x32_bf16(false, a, false, b, (short)0, c, false, false);
}

#ifdef USE_TDM
// TDM: load a (HD x LW x LH x LD) bf16 tile (channels contiguous) from global
// into densely packed LDS rows of HD elements. D# built per ISA 8.3-8.6.
// clang-23 lane: 6-arg builtin (u32x4, i32x8, i32x4, i32x4, i32x8, i32 cpol).
template <int LD, int LH, int LW>
DEV void tdm_load_window(const unsigned short* gsrc, unsigned int lds_byte_addr) {
    unsigned long long ga = (unsigned long long)(const void*)gsrc;
    u32x4 g0;
    g0[0] = 1u;                                            // count=1 (valid), user mode
    g0[1] = lds_byte_addr;                                 // lds_addr
    g0[2] = (unsigned int)(ga & 0xFFFFFFFFu);              // global_addr[31:0]
    g0[3] = (unsigned int)((ga >> 32) & 0x1FFFFFFu)        // global_addr[56:32]
          | (2u << 30);                                    // type=2 ("image")
    i32x8 g1;
    g1[0] = (int)(1u << 16);                               // data_size=1 (2 bytes)
    g1[1] = (int)((HD & 0xFFFF) << 16);                    // tensor_dim0 lo16 (=32)
    g1[2] = (int)(((unsigned)HD >> 16) | ((LW & 0xFFFF) << 16)); // td0 hi, tensor_dim1 lo (=LW)
    g1[3] = (int)((HD & 0xFFFF) << 16);                    // td1 hi=0, tile_dim0=HD
    g1[4] = (int)((LW & 0xFFFF) | ((LH & 0xFFFF) << 16));  // tile_dim1=LW, tile_dim2=LH
    g1[5] = (int)Cc;                                       // tensor_dim0_stride lo32 (=256)
    g1[6] = (int)(((Ww * Cc) & 0xFFFF) << 16);             // s0 hi16=0 ; tensor_dim1_stride lo16
    g1[7] = (int)((unsigned)(Ww * Cc) >> 16);              // tensor_dim1_stride[47:16]
    i32x4 g2;
    g2[0] = LH;                                            // tensor_dim2
    g2[1] = LD;                                            // tensor_dim3
    g2[2] = Hh * Ww * Cc;                                  // tensor_dim2_stride lo32
    g2[3] = (int)((LD & 0xFFFF) << 16);                    // s2 hi16=0 ; tile_dim3=LD
    i32x4 g3 = {0, 0, 0, 0};                               // no dim4
    i32x8 gx = {0, 0, 0, 0, 0, 0, 0, 0};                   // extra operand (clang-23 form)
    __builtin_amdgcn_tensor_load_to_lds(g0, g1, g2, g3, gx, 0);
}
#endif

// ======================= QKV GEMM: qkv[t][row][c] = x @ w_qkv =======================
__global__ __launch_bounds__(256)
void qkv_gemm_kernel(const float* __restrict__ x, const float* __restrict__ w,
                     unsigned short* __restrict__ qkv) {
    constexpr int MT = 128, NT = 64, KT = 32, KDIM = 256, NDIM = 768;
    constexpr int ASTR = KT + 4, BSTR = NT + 8;
    __shared__ unsigned short sA[MT * ASTR];
    __shared__ unsigned short sB[KT * BSTR];

    int tid = threadIdx.x, lane = tid & 31, wave = tid >> 5;
    int rb = blockIdx.y * MT, nb = blockIdx.x * NT;
    v8f acc[4] = {};

    for (int ks = 0; ks < KDIM / KT; ++ks) {
        int kb = ks * KT;
#pragma unroll
        for (int i = 0; i < 4; ++i) {
            int idx = tid + i * 256;
            int r = idx >> 3, c4 = (idx & 7) << 2;
            float4 f = *(const float4*)(x + (size_t)(rb + r) * KDIM + kb + c4);
            unsigned short* dst = sA + r * ASTR + c4;
            dst[0] = f2bf(f.x); dst[1] = f2bf(f.y); dst[2] = f2bf(f.z); dst[3] = f2bf(f.w);
        }
#pragma unroll
        for (int i = 0; i < 2; ++i) {
            int idx = tid + i * 256;
            int r = idx >> 4, c4 = (idx & 15) << 2;
            float4 f = *(const float4*)(w + (size_t)(kb + r) * NDIM + nb + c4);
            unsigned short* dst = sB + r * BSTR + c4;
            dst[0] = f2bf(f.x); dst[1] = f2bf(f.y); dst[2] = f2bf(f.z); dst[3] = f2bf(f.w);
        }
        if (ks + 1 < KDIM / KT)
            __builtin_prefetch(x + (size_t)(rb + (tid >> 3)) * KDIM + kb + KT, 0, 1);
        __syncthreads();

        v16bf a = load_a_frag(sA, wave * 16, 0, ASTR, lane);
#pragma unroll
        for (int nt = 0; nt < 4; ++nt) {
            v16bf b = load_b_frag(sB, 0, nt * 16, BSTR, lane);
            acc[nt] = wmma_bf16(a, b, acc[nt]);
        }
        __syncthreads();
    }
#pragma unroll
    for (int nt = 0; nt < 4; ++nt)
#pragma unroll
        for (int r = 0; r < 8; ++r) {
            int m = wave * 16 + ((lane >> 4) << 3) + r;
            int j = nb + nt * 16 + (lane & 15);
            int t = j >> 8, c = j & 255;
            qkv[((size_t)t * BN + rb + m) * Cc + c] = f2bf(acc[nt][r]);
        }
}

// ======================= Windowed attention (one block = one window+head) =======================
template <int LD, int LH, int LW, int GROUP>
__global__ __launch_bounds__(LD* LH* LW * 2)
void attn_kernel(const unsigned short* __restrict__ qkv,
                 const float* __restrict__ rel_table,
                 unsigned short* __restrict__ attn_out) {
    constexpr int NW = LD * LH * LW;
    constexpr int GD = Dd / LD, GH = Hh / LH, GW = Ww / LW;
    constexpr int NWIN = GD * GH * GW;
    constexpr int KK = NW / 2;             // TOPK_DIV = 2
#ifdef USE_TDM
    constexpr int QSTR = HD;               // TDM packs rows back-to-back
#else
    constexpr int QSTR = HD + 4;
#endif
    constexpr int SSTR = NW + 1;           // padded f32 stride

    __shared__ __align__(16) unsigned short sQ[NW * QSTR];
    __shared__ __align__(16) unsigned short sK[NW * QSTR];
    __shared__ __align__(16) unsigned short sV[NW * QSTR];
    __shared__ float sS[NW * SSTR];

    int tid = threadIdx.x, lane = tid & 31, wave = tid >> 5;
    int bid = blockIdx.x;
    int head  = bid & (GHEADS - 1);
    int wflat = (bid >> 2) % NWIN;
    int b     = (bid >> 2) / NWIN;
    int wd = wflat / (GH * GW);
    int wh = (wflat / GW) % GH;
    int ww = wflat % GW;
    int cbase = GROUP * GDIM + head * HD;

    // ---- stage Q,K,V windows (NW x 32 bf16 each) into LDS ----
#ifdef USE_TDM
    if (wave == 0) {
        size_t n0 = ((size_t)(wd * LD) * Hh + (size_t)(wh * LH)) * Ww + (size_t)(ww * LW);
        size_t base = ((size_t)b * Nn + n0) * Cc + cbase;
        tdm_load_window<LD, LH, LW>(qkv + base,
                                    (unsigned int)(unsigned long long)(const void*)sQ);
        tdm_load_window<LD, LH, LW>(qkv + (size_t)BN * Cc + base,
                                    (unsigned int)(unsigned long long)(const void*)sK);
        tdm_load_window<LD, LH, LW>(qkv + 2 * (size_t)BN * Cc + base,
                                    (unsigned int)(unsigned long long)(const void*)sV);
        __builtin_amdgcn_s_wait_tensorcnt(0);
    }
    __syncthreads();
#else
    for (int idx = tid; idx < NW * HD; idx += blockDim.x) {
        int e = idx >> 5, c = idx & 31;
        int dz = e / (LH * LW), dy = (e / LW) % LH, dx = e % LW;
        size_t n = ((size_t)(wd * LD + dz) * Hh + (wh * LH + dy)) * Ww + (ww * LW + dx);
        size_t base = ((size_t)b * Nn + n) * Cc + cbase + c;
        sQ[e * QSTR + c] = qkv[base];
        sK[e * QSTR + c] = qkv[(size_t)BN * Cc + base];
        sV[e * QSTR + c] = qkv[2 * (size_t)BN * Cc + base];
    }
    __syncthreads();
#endif

    // ---- S = (Q K^T) * scale + rel_bias ; one WMMA per 16x16 tile (K=32=hd) ----
    const float scale = 0.17677669529663687f;  // 32^-0.5
    v16bf aq = load_a_frag(sQ, wave * 16, 0, QSTR, lane);
    for (int j = 0; j < NW / 16; ++j) {
        v16bf bk = load_bT_frag(sK, j * 16, 0, QSTR, lane);
        v8f s = {};
        s = wmma_bf16(aq, bk, s);
#pragma unroll
        for (int r = 0; r < 8; ++r) {
            int m = wave * 16 + ((lane >> 4) << 3) + r;   // query element
            int q = j * 16 + (lane & 15);                 // key element
            int pd = m / (LH * LW), ph = (m / LW) % LH, pw = m % LW;
            int qd = q / (LH * LW), qh = (q / LW) % LH, qw = q % LW;
            int ridx = (pd - qd + LD - 1) * ((2 * LH - 1) * (2 * LW - 1))
                     + (ph - qh + LH - 1) * (2 * LW - 1)
                     + (pw - qw + LW - 1);
            sS[m * SSTR + q] = s[r] * scale + rel_table[ridx * GHEADS + head];
        }
    }
    __syncthreads();

    // ---- per-row top-k threshold (counting select, matches attn >= thr w/ ties) + softmax ----
    if (tid < NW) {
        float* row = sS + tid * SSTR;
        float thr = -3.0e38f;
        for (int i = 0; i < NW; ++i) {
            float e = row[i];
            int cnt = 0;
            for (int jj = 0; jj < NW; ++jj) cnt += (row[jj] >= e) ? 1 : 0;
            if (cnt >= KK && e > thr) thr = e;
        }
        float mx = -3.0e38f;
        for (int i = 0; i < NW; ++i) mx = fmaxf(mx, row[i]);
        float sum = 0.f;
        for (int i = 0; i < NW; ++i) {
            float e = row[i];
            float p = (e >= thr) ? __expf(e - mx) : 0.f;
            row[i] = p;
            sum += p;
        }
        float inv = 1.f / sum;
        for (int i = 0; i < NW; ++i) row[i] *= inv;   // normalized probs, fp32 in LDS
    }
    __syncthreads();

    // ---- O = P @ V  (P gathered fp32->bf16, V bf16) ----
    v8f oacc[2] = {};
    for (int kc = 0; kc < NW / 32; ++kc) {
        v16bf ap = load_a_frag_f32(sS, wave * 16, kc * 32, SSTR, lane);
#pragma unroll
        for (int nt = 0; nt < 2; ++nt) {
            v16bf bv = load_b_frag(sV, kc * 32, nt * 16, QSTR, lane);
            oacc[nt] = wmma_bf16(ap, bv, oacc[nt]);
        }
    }
#pragma unroll
    for (int nt = 0; nt < 2; ++nt)
#pragma unroll
        for (int r = 0; r < 8; ++r) {
            int e = wave * 16 + ((lane >> 4) << 3) + r;
            int c = nt * 16 + (lane & 15);
            int dz = e / (LH * LW), dy = (e / LW) % LH, dx = e % LW;
            size_t n = ((size_t)(wd * LD + dz) * Hh + (wh * LH + dy)) * Ww + (ww * LW + dx);
            attn_out[((size_t)b * Nn + n) * Cc + cbase + c] = f2bf(oacc[nt][r]);
        }
}

// ======================= Output projection: out = attn_out @ w_proj + b =======================
__global__ __launch_bounds__(256)
void proj_gemm_kernel(const unsigned short* __restrict__ a_bf, const float* __restrict__ w,
                      const float* __restrict__ bias, float* __restrict__ out) {
    constexpr int MT = 128, NT = 64, KT = 32, KDIM = 256, NDIM = 256;
    constexpr int ASTR = KT + 4, BSTR = NT + 8;
    __shared__ unsigned short sA[MT * ASTR];
    __shared__ unsigned short sB[KT * BSTR];

    int tid = threadIdx.x, lane = tid & 31, wave = tid >> 5;
    int rb = blockIdx.y * MT, nb = blockIdx.x * NT;
    v8f acc[4] = {};

    for (int ks = 0; ks < KDIM / KT; ++ks) {
        int kb = ks * KT;
#pragma unroll
        for (int i = 0; i < 8; ++i) {
            int idx = tid + i * 256;
            int r = idx >> 4, c2 = (idx & 15) << 1;
            unsigned int v = *(const unsigned int*)(a_bf + (size_t)(rb + r) * KDIM + kb + c2);
            *(unsigned int*)(sA + r * ASTR + c2) = v;
        }
#pragma unroll
        for (int i = 0; i < 2; ++i) {
            int idx = tid + i * 256;
            int r = idx >> 4, c4 = (idx & 15) << 2;
            float4 f = *(const float4*)(w + (size_t)(kb + r) * NDIM + nb + c4);
            unsigned short* dst = sB + r * BSTR + c4;
            dst[0] = f2bf(f.x); dst[1] = f2bf(f.y); dst[2] = f2bf(f.z); dst[3] = f2bf(f.w);
        }
        if (ks + 1 < KDIM / KT)
            __builtin_prefetch(a_bf + (size_t)(rb + (tid >> 4)) * KDIM + kb + KT, 0, 1);
        __syncthreads();

        v16bf a = load_a_frag(sA, wave * 16, 0, ASTR, lane);
#pragma unroll
        for (int nt = 0; nt < 4; ++nt) {
            v16bf b = load_b_frag(sB, 0, nt * 16, BSTR, lane);
            acc[nt] = wmma_bf16(a, b, acc[nt]);
        }
        __syncthreads();
    }
#pragma unroll
    for (int nt = 0; nt < 4; ++nt)
#pragma unroll
        for (int r = 0; r < 8; ++r) {
            int m = wave * 16 + ((lane >> 4) << 3) + r;
            int col = nb + nt * 16 + (lane & 15);
            out[(size_t)(rb + m) * NDIM + col] = acc[nt][r] + bias[col];
        }
}

// ======================= host launcher =======================
extern "C" void kernel_launch(void* const* d_in, const int* in_sizes, int n_in,
                              void* d_out, int out_size, void* d_ws, size_t ws_size,
                              hipStream_t stream) {
    (void)in_sizes; (void)n_in; (void)out_size; (void)ws_size;
    const float* x      = (const float*)d_in[0];
    const float* w_qkv  = (const float*)d_in[1];
    const float* w_proj = (const float*)d_in[2];
    const float* b_proj = (const float*)d_in[3];
    const float* rb0    = (const float*)d_in[4];
    const float* rb1    = (const float*)d_in[5];

    unsigned short* qkv      = (unsigned short*)d_ws;      // 3 * BN * 256 bf16
    unsigned short* attn_out = qkv + (size_t)3 * BN * Cc;  // BN * 256 bf16

    // 1) QKV projection
    qkv_gemm_kernel<<<dim3(768 / 64, BN / 128), 256, 0, stream>>>(x, w_qkv, qkv);

    // 2) windowed attention, group 0: window (2,8,8), Nw=128 -> 2048 blocks, 8 waves
    attn_kernel<2, 8, 8, 0><<<dim3(Bq * (Dd / 2) * (Hh / 8) * (Ww / 8) * GHEADS), 256, 0, stream>>>(
        qkv, rb0, attn_out);

    // 3) windowed attention, group 1: window (4,4,4), Nw=64 -> 4096 blocks, 4 waves
    attn_kernel<4, 4, 4, 1><<<dim3(Bq * (Dd / 4) * (Hh / 4) * (Ww / 4) * GHEADS), 128, 0, stream>>>(
        qkv, rb1, attn_out);

    // 4) output projection + bias
    proj_gemm_kernel<<<dim3(256 / 64, BN / 128), 256, 0, stream>>>(attn_out, w_proj, b_proj,
                                                                   (float*)d_out);
}
